// VectorQuantizer_46222438039688
// MI455X (gfx1250) — matrix operand order repaired and checked by reading, not compiled, not run on hardware
//
#include <hip/hip_runtime.h>

typedef __attribute__((ext_vector_type(16))) __bf16 v16bf;
typedef __attribute__((ext_vector_type(8)))  float  v8f;

#define D_DIM   256
#define K_CODES 1024

__device__ __forceinline__ unsigned short f2bf(float f) {
  unsigned u = __float_as_uint(f);
  u += 0x7FFFu + ((u >> 16) & 1u);          // round-to-nearest-even
  return (unsigned short)(u >> 16);
}

union Frag { v16bf v; unsigned u[8]; };

// ---------------- Kernel 1: codebook -> bf16, ||e||^2 (exact f32) ----------
__global__ __launch_bounds__(256) void vq_prep(const float* __restrict__ cb,
                                               unsigned short* __restrict__ cbbf,
                                               float* __restrict__ enorm2) {
  const int k = blockIdx.x;           // one code per block
  const int d = threadIdx.x;          // one dim per thread (256)
  float v = cb[(size_t)k * D_DIM + d];
  cbbf[(size_t)k * D_DIM + d] = f2bf(v);
  float sq = v * v;
  #pragma unroll
  for (int off = 16; off; off >>= 1) sq += __shfl_xor(sq, off, 32);
  __shared__ float sm[8];
  if ((threadIdx.x & 31) == 0) sm[threadIdx.x >> 5] = sq;
  __syncthreads();
  if (threadIdx.x == 0) {
    float t = 0.f;
    #pragma unroll
    for (int i = 0; i < 8; ++i) t += sm[i];
    enorm2[k] = t;
  }
}

// ---- Kernel 2: fused GEMM (WMMA bf16, M=32 per wave) + argmin + outputs ---
__global__ __launch_bounds__(256) void vq_main(const float* __restrict__ x,
                                               const float* __restrict__ cb,
                                               const unsigned short* __restrict__ cbbf,
                                               const float* __restrict__ enorm2,
                                               float* __restrict__ qout,
                                               float* __restrict__ idxout,
                                               float* __restrict__ rowcommit) {
  const int lane = threadIdx.x & 31;
  const int wave = threadIdx.x >> 5;
  const int m16  = lane & 15;
  const int hi   = lane >> 4;
  const int row0 = blockIdx.x * 256 + wave * 32;   // 8 waves x 32 rows

  // ---- Build A fragments for two 16-row tiles (ISA 16-bit A 16x32 layout);
  //      accumulate exact f32 ||x||^2 partials (lane holds 128 dims of its row).
  Frag  a[2][8];
  float xfull[2];
  #pragma unroll
  for (int t = 0; t < 2; ++t) {
    float xpart = 0.f;
    const float* xrow = x + (size_t)(row0 + t * 16 + m16) * D_DIM;
    #pragma unroll
    for (int ks = 0; ks < 8; ++ks) {
      const int kb = ks * 32 + (hi ? 8 : 0);
      #pragma unroll
      for (int r = 0; r < 8; ++r) {
        const int k = kb + (r < 4 ? 2 * r : 16 + 2 * (r - 4));
        const float f0 = xrow[k], f1 = xrow[k + 1];
        xpart = __builtin_fmaf(f0, f0, xpart);
        xpart = __builtin_fmaf(f1, f1, xpart);
        a[t][ks].u[r] = (unsigned)f2bf(f0) | ((unsigned)f2bf(f1) << 16);
      }
    }
    xfull[t] = xpart + __shfl_xor(xpart, 16, 32);  // lanes l and l+16 complement
  }

  float bv[2][8];
  int   bi[2][8];
  #pragma unroll
  for (int t = 0; t < 2; ++t)
    #pragma unroll
    for (int r = 0; r < 8; ++r) { bv[t][r] = 3.4e38f; bi[t][r] = 0; }

  // ---- Sweep all 64 code tiles; each B fragment feeds both row tiles
  for (int c0 = 0; c0 < K_CODES; c0 += 16) {
    v8f acc0 = {0.f, 0.f, 0.f, 0.f, 0.f, 0.f, 0.f, 0.f};
    v8f acc1 = {0.f, 0.f, 0.f, 0.f, 0.f, 0.f, 0.f, 0.f};
    // B layout (32x16 K-major): lanes 0-15 hold K=0..7,16..23, lanes 16-31 hold
    // K=8..15,24..31 (same order as A); lane&15 selects the code column.
    const unsigned short* brow = cbbf + (size_t)(c0 + m16) * D_DIM + (hi ? 8 : 0);
    #pragma unroll
    for (int ks = 0; ks < 8; ++ks) {
      Frag b;
      #pragma unroll
      for (int r = 0; r < 8; ++r)
        b.u[r] = *(const unsigned*)(brow + ks * 32 + (r < 4 ? 2 * r : 16 + 2 * (r - 4)));
      acc0 = __builtin_amdgcn_wmma_f32_16x16x32_bf16(
          false, a[0][ks].v, false, b.v, (short)0, acc0, false, false);
      acc1 = __builtin_amdgcn_wmma_f32_16x16x32_bf16(
          false, a[1][ks].v, false, b.v, (short)0, acc1, false, false);
    }
    const float en   = enorm2[c0 + m16];
    const int   code = c0 + m16;
    #pragma unroll
    for (int r = 0; r < 8; ++r) {
      const float s0 = __builtin_fmaf(-2.f, acc0[r], en);  // ||e||^2 - 2 x.e
      const float s1 = __builtin_fmaf(-2.f, acc1[r], en);
      if (s0 < bv[0][r]) { bv[0][r] = s0; bi[0][r] = code; }
      if (s1 < bv[1][r]) { bv[1][r] = s1; bi[1][r] = code; }
    }
  }

  // ---- Cross-lane argmin within each 16-lane half (first-min tie break)
  #pragma unroll
  for (int t = 0; t < 2; ++t)
    #pragma unroll
    for (int r = 0; r < 8; ++r) {
      float v = bv[t][r]; int i = bi[t][r];
      #pragma unroll
      for (int off = 1; off < 16; off <<= 1) {
        const float ov = __shfl_xor(v, off, 32);
        const int   oi = __shfl_xor(i, off, 32);
        if (ov < v || (ov == v && oi < i)) { v = ov; i = oi; }
      }
      bv[t][r] = v; bi[t][r] = i;
    }

  // ---- Outputs: indices, per-row commitment, quantized rows (float4 gather)
  #pragma unroll
  for (int t = 0; t < 2; ++t)
    #pragma unroll
    for (int r = 0; r < 8; ++r) {
      const int rowl = r + (hi ? 8 : 0);
      const int rowg = row0 + t * 16 + rowl;
      const float xn = __shfl(xfull[t], rowl, 32);   // row's exact ||x||^2
      if (m16 == 0) {
        idxout[rowg]    = (float)bi[t][r];
        rowcommit[rowg] = xn + bv[t][r];             // ||x-e||^2 for this row
      }
      const float4* src = (const float4*)(cb   + (size_t)bi[t][r] * D_DIM);
      float4*       dst = (float4*)      (qout + (size_t)rowg     * D_DIM);
      #pragma unroll
      for (int j = 0; j < 4; ++j)
        dst[j * 16 + m16] = src[j * 16 + m16];
    }
}

// ---------------- Kernel 3: deterministic loss reduction -------------------
__global__ __launch_bounds__(256) void vq_reduce(const float* __restrict__ rc,
                                                 float* __restrict__ loss, int N) {
  float s = 0.f;
  for (int i = threadIdx.x; i < N; i += 256) s += rc[i];
  #pragma unroll
  for (int off = 16; off; off >>= 1) s += __shfl_xor(s, off, 32);
  __shared__ float sm[8];
  if ((threadIdx.x & 31) == 0) sm[threadIdx.x >> 5] = s;
  __syncthreads();
  if (threadIdx.x == 0) {
    float t = 0.f;
    #pragma unroll
    for (int i = 0; i < 8; ++i) t += sm[i];
    *loss = t / ((float)N * (float)D_DIM);
  }
}

extern "C" void kernel_launch(void* const* d_in, const int* in_sizes, int n_in,
                              void* d_out, int out_size, void* d_ws, size_t ws_size,
                              hipStream_t stream) {
  const float* x  = (const float*)d_in[0];   // (16, 2048, 256) f32
  const float* cb = (const float*)d_in[1];   // (1024, 256) f32
  const int N = in_sizes[0] / D_DIM;         // 32768 rows

  // Workspace layout: bf16 codebook | ||e||^2 | per-row commitment
  unsigned short* cbbf   = (unsigned short*)d_ws;
  float*          enorm2 = (float*)((char*)d_ws + (size_t)K_CODES * D_DIM * sizeof(unsigned short));
  float*          rcomm  = enorm2 + K_CODES;

  float* qout   = (float*)d_out;                  // N*256 quantized
  float* idxout = qout + (size_t)N * D_DIM;       // N indices (as float)
  float* loss   = idxout + N;                     // 1 scalar

  vq_prep  <<<K_CODES, 256, 0, stream>>>(cb, cbbf, enorm2);
  vq_main  <<<N / 256, 256, 0, stream>>>(x, cb, cbbf, enorm2, qout, idxout, rcomm);
  vq_reduce<<<1,       256, 0, stream>>>(rcomm, loss, N);
}
